// RNNWithAttention_28991029248777
// MI455X (gfx1250) — compile-verified, matrix operand order checked by reading, not run on hardware
//
#include <hip/hip_runtime.h>
#include <hip/hip_bf16.h>
#include <math.h>

// Model dims (fixed by the reference)
#define VV 32000
#define EE 32
#define HH 128
#define BB 2
#define TT 512
#define BT (BB * TT)

typedef __attribute__((ext_vector_type(16))) _Float16 v16h;
typedef __attribute__((ext_vector_type(8)))  _Float16 v8h;
typedef __attribute__((ext_vector_type(8)))  float    v8f;

// ---------------------------------------------------------------------------
// f32 -> f16 elementwise convert
// ---------------------------------------------------------------------------
__global__ void k_f32_to_f16(const float* __restrict__ src, _Float16* __restrict__ dst, int n) {
    int i = blockIdx.x * blockDim.x + threadIdx.x;
    if (i < n) dst[i] = (_Float16)src[i];
}

// ---------------------------------------------------------------------------
// Embedding lookup + input projection:
// xp[row,h] = sum_e embed[x[row],e] * W_ih[h,e] + b_ih[h] + b_hh[h]
// grid = BT blocks, 128 threads
// ---------------------------------------------------------------------------
__global__ void k_embed_proj(const int* __restrict__ x, const float* __restrict__ tbl,
                             const float* __restrict__ Wih, const float* __restrict__ bih,
                             const float* __restrict__ bhh, float* __restrict__ xp) {
    int row = blockIdx.x;
    int h   = threadIdx.x;
    int tok = x[row];
    const float* e = tbl + (long long)tok * EE;
    const float* w = Wih + h * EE;
    float acc = bih[h] + bhh[h];
#pragma unroll
    for (int j = 0; j < EE; ++j) acc += e[j] * w[j];
    xp[row * HH + h] = acc;
}

// ---------------------------------------------------------------------------
// Sequential tanh RNN — latency-optimized.
// One block per batch, 512 threads: 4 threads per output element (split-K),
// each holding its 32-float W_hh slice in registers. Partials combined with
// wave32 shuffles; h lives in LDS; xp load software-pipelined one step ahead.
// ---------------------------------------------------------------------------
__global__ void k_rnn(const float* __restrict__ xp, const float* __restrict__ h0,
                      const float* __restrict__ Whh, float* __restrict__ rnn_out,
                      float* __restrict__ hidden_out) {
    __shared__ float h[HH];
    int b   = blockIdx.x;
    int tid = threadIdx.x;   // 0..511
    int i   = tid >> 2;      // output element 0..127
    int p   = tid & 3;       // split-K partial 0..3 (adjacent lanes of a wave)

    // Register-resident W_hh slice: W[i][p*32 .. p*32+31]
    float wreg[32];
    const float* wr = Whh + i * HH + p * 32;
#pragma unroll
    for (int j = 0; j < 32; ++j) wreg[j] = wr[j];

    if (tid < HH) h[tid] = h0[b * HH + tid];
    __syncthreads();

    const float* xrow = xp + (long long)b * TT * HH;
    float xv = xrow[i];  // prefetch step 0
    for (int t = 0; t < TT; ++t) {
        float acc = 0.0f;
        const float* hp = h + p * 32;
#pragma unroll
        for (int j = 0; j < 32; ++j) acc += wreg[j] * hp[j];
        // combine 4 partials within the wave (lanes differ only in p)
        acc += __shfl_xor(acc, 1);
        acc += __shfl_xor(acc, 2);
        float xcur = xv;
        if (t + 1 < TT) xv = xrow[(t + 1) * HH + i];  // prefetch next step
        float hv = tanhf(acc + xcur);
        __syncthreads();  // all reads of h complete before overwrite
        if (p == 0) {
            h[i] = hv;
            rnn_out[(long long)(b * TT + t) * HH + i] = hv;
        }
        __syncthreads();  // new h visible
    }
    if (tid < HH) hidden_out[b * HH + tid] = h[tid];
}

// ---------------------------------------------------------------------------
// WMMA f16 GEMM:  C[m,n] = sum_k A[m,k] * W[n,k] (+ bias[n]) (opt relu)
// A: M x K f16 row-major, W: N x K f16 row-major, C: M x N f32.
// One wave32 keeps the A fragments (K/32 x v16h) in registers and sweeps NT
// consecutive 16-wide N tiles, chaining K/32 v_wmma_f32_16x16x32_f16 each.
// ---------------------------------------------------------------------------
__device__ __forceinline__ v16h load_frag(const _Float16* __restrict__ base, int ld,
                                          int row, int k0, int hsel) {
    const _Float16* p = base + (long long)row * ld + k0 + hsel * 8;
    v8h lo = *(const v8h*)(p);        // K = k0 + hsel*8      .. +7
    v8h hi = *(const v8h*)(p + 16);   // K = k0 + 16 + hsel*8 .. +7
    return __builtin_shufflevector(lo, hi, 0, 1, 2, 3, 4, 5, 6, 7,
                                           8, 9, 10, 11, 12, 13, 14, 15);
}

template <int K, int NT, int RELU>
__global__ void k_gemm_wmma(const _Float16* __restrict__ A, const _Float16* __restrict__ W,
                            const float* __restrict__ bias, float* __restrict__ C,
                            int M, int N) {
    int wid  = (blockIdx.x * blockDim.x + threadIdx.x) >> 5;  // wave-uniform
    int lane = threadIdx.x & 31;
    int groups_n = (N >> 4) / NT;
    int tiles_m  = M >> 4;
    if (wid >= tiles_m * groups_n) return;  // whole wave exits: EXEC all-1s for WMMA
    int tm = wid / groups_n;
    int gn = wid - tm * groups_n;

    int nloc = lane & 15;   // = m for A fragment, = n for B fragment / C columns
    int hsel = lane >> 4;

    // A fragments resident in registers, reused across NT N-tiles
    v16h a[K / 32];
#pragma unroll
    for (int kc = 0; kc < K / 32; ++kc)
        a[kc] = load_frag(A, K, tm * 16 + nloc, kc * 32, hsel);

#pragma unroll
    for (int u = 0; u < NT; ++u) {
        int tn = gn * NT + u;
        v8f c;
        float bv = bias ? bias[tn * 16 + nloc] : 0.0f;
#pragma unroll
        for (int r = 0; r < 8; ++r) c[r] = bv;

#pragma unroll
        for (int kc = 0; kc < K / 32; ++kc) {
            v16h bfrag = load_frag(W, K, tn * 16 + nloc, kc * 32, hsel);
            c = __builtin_amdgcn_wmma_f32_16x16x32_f16(false, a[kc], false, bfrag,
                                                       (short)0, c, false, false);
        }

        // C/D layout: VGPR r, lane L -> row m = r + 8*(L>>4), col n = L&15
        float* out = C + (long long)(tm * 16 + hsel * 8) * N + tn * 16 + nloc;
#pragma unroll
        for (int r = 0; r < 8; ++r) {
            float vv = c[r];
            if (RELU) vv = fmaxf(vv, 0.0f);
            out[(long long)r * N] = vv;
        }
    }
}

// ---------------------------------------------------------------------------
// Additive attention scores + causal softmax.
// One block per (b,t), 256 threads. scores[s] = sum_h v[h]*tanh(q[b,t,h]+k[b,s,h])
// ---------------------------------------------------------------------------
__global__ void k_attn(const float* __restrict__ q, const float* __restrict__ kk,
                       const float* __restrict__ vvec, float* __restrict__ wts) {
    __shared__ float qs[HH];
    __shared__ float vs[HH];
    __shared__ float sc[TT];
    __shared__ float red[256];
    int b = blockIdx.x / TT;
    int t = blockIdx.x - b * TT;
    int tid = threadIdx.x;

    if (tid < HH) {
        qs[tid] = q[(long long)(b * TT + t) * HH + tid];
        vs[tid] = vvec[tid];
    }
    __syncthreads();

    for (int s = tid; s <= t; s += 256) {
        const float* kr = kk + (long long)(b * TT + s) * HH;
        float acc = 0.0f;
#pragma unroll 4
        for (int hh = 0; hh < HH; ++hh) acc += vs[hh] * tanhf(qs[hh] + kr[hh]);
        sc[s] = acc;
    }
    __syncthreads();

    // max-reduce over s in [0, t]
    float m = -3.4e38f;
    for (int s = tid; s <= t; s += 256) m = fmaxf(m, sc[s]);
    red[tid] = m;
    __syncthreads();
    for (int off = 128; off > 0; off >>= 1) {
        if (tid < off) red[tid] = fmaxf(red[tid], red[tid + off]);
        __syncthreads();
    }
    m = red[0];
    __syncthreads();

    // exp + sum-reduce
    float sum = 0.0f;
    for (int s = tid; s <= t; s += 256) {
        float e = __expf(sc[s] - m);
        sc[s] = e;
        sum += e;
    }
    red[tid] = sum;
    __syncthreads();
    for (int off = 128; off > 0; off >>= 1) {
        if (tid < off) red[tid] += red[tid + off];
        __syncthreads();
    }
    float inv = 1.0f / red[0];

    float* wrow = wts + (long long)(b * TT + t) * TT;
    for (int s = tid; s < TT; s += 256) wrow[s] = (s <= t) ? sc[s] * inv : 0.0f;
}

// ---------------------------------------------------------------------------
// context[b,t,h] = sum_{s<=t} weights[b,t,s] * rnn[b,s,h]; emit combined
// = concat(rnn, context) directly in f16 for the Wp GEMM.
// ---------------------------------------------------------------------------
__global__ void k_context(const float* __restrict__ wts, const float* __restrict__ rnn,
                          _Float16* __restrict__ comb) {
    int b = blockIdx.x / TT;
    int t = blockIdx.x - b * TT;
    int hh = threadIdx.x;
    const float* wrow = wts + (long long)(b * TT + t) * TT;
    float acc = 0.0f;
    for (int s = 0; s <= t; ++s)
        acc += wrow[s] * rnn[(long long)(b * TT + s) * HH + hh];
    long long r = (long long)(b * TT + t);
    comb[r * (2 * HH) + hh]      = (_Float16)rnn[r * HH + hh];
    comb[r * (2 * HH) + HH + hh] = (_Float16)acc;
}

// ---------------------------------------------------------------------------
// launch
// ---------------------------------------------------------------------------
extern "C" void kernel_launch(void* const* d_in, const int* in_sizes, int n_in,
                              void* d_out, int out_size, void* d_ws, size_t ws_size,
                              hipStream_t stream) {
    (void)in_sizes; (void)n_in; (void)out_size; (void)ws_size;

    const int*   x    = (const int*)  d_in[0];
    const float* h0   = (const float*)d_in[1];
    const float* tbl  = (const float*)d_in[2];
    const float* Wih  = (const float*)d_in[3];
    const float* Whh  = (const float*)d_in[4];
    const float* bih  = (const float*)d_in[5];
    const float* bhh  = (const float*)d_in[6];
    const float* W1   = (const float*)d_in[7];
    const float* W2   = (const float*)d_in[8];
    const float* vvec = (const float*)d_in[9];
    const float* Wp   = (const float*)d_in[10];
    const float* bp   = (const float*)d_in[11];
    const float* Wfc  = (const float*)d_in[12];
    const float* bfc  = (const float*)d_in[13];

    // Output layout: logits (B,T,V) | hidden (1,B,H) | weights (B,T,T)
    float* logits     = (float*)d_out;
    float* hidden_out = logits + (long long)BT * VV;
    float* wts        = hidden_out + BB * HH;

    // Workspace carve-up
    char* w = (char*)d_ws;
    size_t off = 0;
    auto take = [&](size_t bytes) { char* p = w + off; off += (bytes + 255) & ~(size_t)255; return p; };
    float*    xp     = (float*)   take((size_t)BT * HH * 4);
    float*    rnn    = (float*)   take((size_t)BT * HH * 4);
    float*    q      = (float*)   take((size_t)BT * HH * 4);
    float*    kproj  = (float*)   take((size_t)BT * HH * 4);
    float*    outf   = (float*)   take((size_t)BT * HH * 4);
    _Float16* rnn16  = (_Float16*)take((size_t)BT * HH * 2);
    _Float16* comb16 = (_Float16*)take((size_t)BT * 2 * HH * 2);
    _Float16* out16  = (_Float16*)take((size_t)BT * HH * 2);
    _Float16* W1h    = (_Float16*)take((size_t)HH * HH * 2);
    _Float16* W2h    = (_Float16*)take((size_t)HH * HH * 2);
    _Float16* Wph    = (_Float16*)take((size_t)HH * 2 * HH * 2);
    _Float16* Wfch   = (_Float16*)take((size_t)VV * HH * 2);

    auto cvt = [&](const float* s, _Float16* d, int n) {
        k_f32_to_f16<<<(n + 255) / 256, 256, 0, stream>>>(s, d, n);
    };

    // Weight conversions (independent of activations)
    cvt(W1,  W1h,  HH * HH);
    cvt(W2,  W2h,  HH * HH);
    cvt(Wp,  Wph,  HH * 2 * HH);
    cvt(Wfc, Wfch, VV * HH);

    // 1) embedding + input projection
    k_embed_proj<<<BT, HH, 0, stream>>>(x, tbl, Wih, bih, bhh, xp);

    // 2) sequential RNN, split-K latency-optimized (also writes hidden_out)
    k_rnn<<<BB, 512, 0, stream>>>(xp, h0, Whh, rnn, hidden_out);
    cvt(rnn, rnn16, BT * HH);

    // 3) query / keys GEMMs (WMMA): M=1024, N=128, K=128, NT=4 -> 128 waves
    {
        int waves  = (BT / 16) * ((HH / 16) / 4);
        int blocks = (waves + 3) / 4;
        k_gemm_wmma<HH, 4, 0><<<blocks, 128, 0, stream>>>(rnn16, W1h, nullptr, q, BT, HH);
        k_gemm_wmma<HH, 4, 0><<<blocks, 128, 0, stream>>>(rnn16, W2h, nullptr, kproj, BT, HH);
    }

    // 4) additive attention + causal softmax (writes weights output)
    k_attn<<<BT, 256, 0, stream>>>(q, kproj, vvec, wts);

    // 5) context + concat -> combined f16
    k_context<<<BT, HH, 0, stream>>>(wts, rnn, comb16);

    // 6) out = relu(combined @ Wp^T + bp): M=1024, N=128, K=256, NT=4
    {
        int waves  = (BT / 16) * ((HH / 16) / 4);
        int blocks = (waves + 3) / 4;
        k_gemm_wmma<2 * HH, 4, 1><<<blocks, 128, 0, stream>>>(comb16, Wph, bp, outf, BT, HH);
    }
    cvt(outf, out16, BT * HH);

    // 7) logits = out @ Wfc^T + bfc: M=1024, N=32000, K=128, NT=4 -> 32000 waves
    {
        int waves  = (BT / 16) * ((VV / 16) / 4);
        int blocks = (waves + 3) / 4;
        k_gemm_wmma<HH, 4, 0><<<blocks, 128, 0, stream>>>(out16, Wfch, bfc, logits, BT, VV);
    }
}
